// Conv2dImg2Col_38276748542860
// MI455X (gfx1250) — matrix operand order repaired and checked by reading, not compiled, not run on hardware
//
#include <hip/hip_runtime.h>

// ---------------------------------------------------------------------------
// Conv2d 3x3 s1 p1 (NCHW/OIHW) as implicit GEMM, bf16 WMMA / f32 accumulate.
// Fast path (needs ~26.3MB scratch):
//   1) transform x  -> channels-last bf16  xbf[n][h][w][ci]   (im2col rows contiguous)
//   2) transform w  -> tap-major bf16      wbf[co][tap][ci]   (k = tap*128+ci)
//   3) main kernel: block tile 128(M)x128(N), 128 thr (4 wave32), wave tile 32x128.
//      A and B fragments loaded DIRECTLY from global as b128 (no LDS, no barriers);
//      all 8 B fragments of a K-step kept in flight so load latency overlaps WMMA.
//      Padding handled by a zero guard row in scratch (pointer select per tap).
// Fallback path: LDS-staged kernel (BN=64).
// ---------------------------------------------------------------------------

typedef __attribute__((ext_vector_type(16))) __bf16        v16bf;
typedef __attribute__((ext_vector_type(8)))  float         v8f;
typedef __attribute__((ext_vector_type(4)))  unsigned int  v4u;

#define CIN   128
#define HH    56
#define WW    56
#define COUT  256
#define HWSZ  3136
#define KDIM  1152                 // 9 * 128, tap-major
#define NIMG  32

union AFrag  { v16bf v; v4u q[2]; };
union BFrag  { v16bf v; v4u q[2]; };
union CFrag  { v8f  v; float f[8]; };
union Pack16 { __bf16 h[16]; v4u q[2]; };

// ---------------- scratch layout ----------------
#define GUARD_BYTES 512                                   // zero row (256B used)
#define XBF_ELEMS   ((size_t)NIMG * HWSZ * CIN)           // 12,845,056
#define WBF_ELEMS   ((size_t)COUT * KDIM)                 //    294,912
#define WS_NEEDED   (GUARD_BYTES + (XBF_ELEMS + WBF_ELEMS) * 2)

// ===========================================================================
// Transform kernels
// ===========================================================================
__global__ __launch_bounds__(256)
void transform_w(const float* __restrict__ wgt, unsigned short* __restrict__ ws0)
{
    // zero guard row
    if (blockIdx.x == 0 && threadIdx.x < 128)
        ((unsigned int*)ws0)[threadIdx.x] = 0u;

    unsigned short* wbf = ws0 + GUARD_BYTES / 2 + XBF_ELEMS;
    const int idx = blockIdx.x * 256 + threadIdx.x;       // over COUT*KDIM
    if (idx >= (int)WBF_ELEMS) return;
    const int co  = idx / KDIM;
    const int rk  = idx - co * KDIM;
    const int tap = rk >> 7;                              // /128
    const int ci  = rk & 127;
    const float v = wgt[co * KDIM + ci * 9 + tap];
    __bf16 b = (__bf16)v;
    wbf[idx] = __builtin_bit_cast(unsigned short, b);
}

__global__ __launch_bounds__(256)
void transform_x(const float* __restrict__ x, unsigned short* __restrict__ ws0)
{
    unsigned short* xbf = ws0 + GUARD_BYTES / 2;
    const size_t idx = (size_t)blockIdx.x * 256 + threadIdx.x;  // over N*HW*CIN
    if (idx >= XBF_ELEMS) return;
    const int ci  = (int)(idx & 127);
    const size_t rest = idx >> 7;                         // n*HW + p
    const int n  = (int)(rest / HWSZ);
    const int p  = (int)(rest - (size_t)n * HWSZ);
    const float v = x[((size_t)n * CIN + ci) * HWSZ + p];
    __bf16 b = (__bf16)v;
    xbf[idx] = __builtin_bit_cast(unsigned short, b);
}

// ===========================================================================
// Fast main kernel: BM=128, BN=128, 128 threads, load-direct fragments
// ===========================================================================
__global__ __launch_bounds__(128)
void conv3x3_wmma_direct(const unsigned short* __restrict__ ws0,
                         const float* __restrict__ bias,
                         float* __restrict__ out)
{
    const unsigned short* __restrict__ zerog = ws0;                    // 256B zeros
    const unsigned short* __restrict__ xbf   = ws0 + GUARD_BYTES / 2;
    const unsigned short* __restrict__ wbf   = xbf + XBF_ELEMS;

    const int tid    = threadIdx.x;
    const int wave   = tid >> 5;
    const int lane   = tid & 31;
    const int lane15 = lane & 15;
    const int khalf  = lane >> 4;           // 0/1
    const int kh8    = khalf * 8;           // A k offset = 2*kh8 (bf16 elems)
    const int nblk   = blockIdx.x;          // 0..1   (C_OUT tiles of 128)
    const int mblk   = blockIdx.y;          // 0..783 (pixel tiles of 128)

    // per-lane A row pixels for the two M-fragments
    int ohv[2], owv[2];
    size_t imgBase[2];
#pragma unroll
    for (int i = 0; i < 2; ++i) {
        const int mloc = wave * 32 + i * 16 + lane15;
        const int gm   = mblk * 128 + mloc;
        const int nImg = gm / HWSZ;
        const int rem  = gm - nImg * HWSZ;
        ohv[i] = rem / WW;
        owv[i] = rem - ohv[i] * WW;
        imgBase[i] = (size_t)nImg * HWSZ;
    }

    CFrag acc[2][8];
#pragma unroll
    for (int i = 0; i < 2; ++i)
#pragma unroll
        for (int j = 0; j < 8; ++j) { v8f z = {}; acc[i][j].v = z; }

    // per-j B base pointers (co fixed per lane): wbf[co][tap*128 + ci]
    const unsigned short* __restrict__ bbase[8];
#pragma unroll
    for (int j = 0; j < 8; ++j) {
        const int co = nblk * 128 + j * 16 + lane15;
        bbase[j] = wbf + (size_t)co * KDIM + khalf * 16;
    }

#pragma unroll 1
    for (int tap = 0; tap < 9; ++tap) {
        const int r = tap / 3;
        const int s = tap - r * 3;

        // A row pointers for this tap (zero guard when padded out)
        const unsigned short* arow[2];
#pragma unroll
        for (int i = 0; i < 2; ++i) {
            const int ih = ohv[i] + r - 1;
            const int iw = owv[i] + s - 1;
            const bool ok = ((unsigned)ih < (unsigned)HH) & ((unsigned)iw < (unsigned)WW);
            arow[i] = ok ? (xbf + (imgBase[i] + ih * WW + iw) * CIN) : zerog;
        }

#pragma unroll
        for (int cc = 0; cc < 4; ++cc) {
            // A frags: elems 0..15 -> k=16*khalf+0..15
            AFrag af[2];
#pragma unroll
            for (int i = 0; i < 2; ++i) {
                const unsigned short* ap = arow[i] + cc * 32;
                af[i].q[0] = *(const v4u*)(ap + 2 * kh8);
                af[i].q[1] = *(const v4u*)(ap + 2 * kh8 + 8);
            }
            // ALL 8 B frags issued before any WMMA -> 16 loads in flight,
            // waits staggered across the 16 WMMAs below.
            BFrag b[8];
#pragma unroll
            for (int j = 0; j < 8; ++j) {
                const unsigned short* bp = bbase[j] + tap * 128 + cc * 32;
                b[j].q[0] = *(const v4u*)bp;
                b[j].q[1] = *(const v4u*)(bp + 8);
            }
#pragma unroll
            for (int j = 0; j < 8; ++j) {
                acc[0][j].v = __builtin_amdgcn_wmma_f32_16x16x32_bf16(
                    false, af[0].v, false, b[j].v, (short)0, acc[0][j].v, false, false);
                acc[1][j].v = __builtin_amdgcn_wmma_f32_16x16x32_bf16(
                    false, af[1].v, false, b[j].v, (short)0, acc[1][j].v, false, false);
            }
        }
    }

    // epilogue: bias + scatter store (NCHW)
    const int mAdd = khalf * 8;
#pragma unroll
    for (int j = 0; j < 8; ++j) {
        const int co = nblk * 128 + j * 16 + lane15;
        const float bv = bias[co];
#pragma unroll
        for (int i = 0; i < 2; ++i) {
#pragma unroll
            for (int rr = 0; rr < 8; ++rr) {
                const int mloc = wave * 32 + i * 16 + rr + mAdd;
                const int g    = mblk * 128 + mloc;
                const int ni   = g / HWSZ;
                const int prem = g - ni * HWSZ;
                out[((size_t)ni * COUT + co) * HWSZ + prem] = acc[i][j].f[rr] + bv;
            }
        }
    }
}

// ===========================================================================
// Fallback kernel (LDS staged, BN=64) — used if scratch too small
// ===========================================================================
#define A_STRIDE 136
#define B_STRIDE 136

__global__ __launch_bounds__(128)
void conv3x3_wmma_lds(const float* __restrict__ x,
                      const float* __restrict__ wgt,
                      const float* __restrict__ bias,
                      float* __restrict__ out)
{
    __shared__ unsigned short Ash[128 * A_STRIDE];
    __shared__ unsigned short Bsh[64 * B_STRIDE];

    const int tid    = threadIdx.x;
    const int wave   = tid >> 5;
    const int lane   = tid & 31;
    const int lane15 = lane & 15;
    const int khalf  = lane >> 4;
    const int kh8    = khalf * 8;
    const int nblk   = blockIdx.x;
    const int mblk   = blockIdx.y;

    const int gm   = mblk * 128 + tid;
    const int nImg = gm / HWSZ;
    const int rem  = gm - nImg * HWSZ;
    const int oh   = rem / WW;
    const int ow   = rem - oh * WW;
    const float* __restrict__ xn = x + (size_t)nImg * CIN * HWSZ;

    const int bco     = tid & 63;
    const int bciBase = (tid >> 6) * 64;
    const float* __restrict__ wrow =
        wgt + (size_t)(nblk * 64 + bco) * (CIN * 9) + (size_t)bciBase * 9;

    CFrag acc[2][4];
#pragma unroll
    for (int i = 0; i < 2; ++i)
#pragma unroll
        for (int j = 0; j < 4; ++j) { v8f z = {}; acc[i][j].v = z; }

#pragma unroll 1
    for (int tap = 0; tap < 9; ++tap) {
        const int r  = tap / 3;
        const int s  = tap - r * 3;
        const int ih = oh + r - 1;
        const int iw = ow + s - 1;
        const bool ok = ((unsigned)ih < (unsigned)HH) & ((unsigned)iw < (unsigned)WW);
        const int base = ok ? (ih * WW + iw) : 0;

#pragma unroll
        for (int c0 = 0; c0 < CIN; c0 += 16) {
            Pack16 p;
#pragma unroll
            for (int e = 0; e < 16; ++e) {
                const float v = xn[(c0 + e) * HWSZ + base];
                p.h[e] = (__bf16)(ok ? v : 0.0f);
            }
            v4u* dst = (v4u*)&Ash[tid * A_STRIDE + c0];
            dst[0] = p.q[0]; dst[1] = p.q[1];
        }
#pragma unroll
        for (int c0 = 0; c0 < 64; c0 += 16) {
            Pack16 p;
#pragma unroll
            for (int e = 0; e < 16; ++e)
                p.h[e] = (__bf16)wrow[(c0 + e) * 9 + tap];
            v4u* dst = (v4u*)&Bsh[bco * B_STRIDE + bciBase + c0];
            dst[0] = p.q[0]; dst[1] = p.q[1];
        }
        __syncthreads();

#pragma unroll
        for (int cc = 0; cc < 4; ++cc) {
            AFrag af[2];
#pragma unroll
            for (int i = 0; i < 2; ++i) {
                const int mloc = wave * 32 + i * 16 + lane15;
                const unsigned short* ap = &Ash[mloc * A_STRIDE + cc * 32];
                af[i].q[0] = *(const v4u*)(ap + 2 * kh8);
                af[i].q[1] = *(const v4u*)(ap + 2 * kh8 + 8);
            }
            BFrag bfr[4];
#pragma unroll
            for (int j = 0; j < 4; ++j) {
                const unsigned short* bp =
                    &Bsh[(j * 16 + lane15) * B_STRIDE + cc * 32 + khalf * 16];
                bfr[j].q[0] = *(const v4u*)bp;
                bfr[j].q[1] = *(const v4u*)(bp + 8);
            }
#pragma unroll
            for (int i = 0; i < 2; ++i)
#pragma unroll
                for (int j = 0; j < 4; ++j)
                    acc[i][j].v = __builtin_amdgcn_wmma_f32_16x16x32_bf16(
                        false, af[i].v, false, bfr[j].v, (short)0, acc[i][j].v,
                        false, false);
        }
        __syncthreads();
    }

    const int mAdd = khalf * 8;
#pragma unroll
    for (int j = 0; j < 4; ++j) {
        const int co = nblk * 64 + j * 16 + lane15;
        const float bv = bias[co];
#pragma unroll
        for (int i = 0; i < 2; ++i) {
#pragma unroll
            for (int rr = 0; rr < 8; ++rr) {
                const int mloc = wave * 32 + i * 16 + rr + mAdd;
                const int g    = mblk * 128 + mloc;
                const int ni   = g / HWSZ;
                const int prem = g - ni * HWSZ;
                out[((size_t)ni * COUT + co) * HWSZ + prem] = acc[i][j].f[rr] + bv;
            }
        }
    }
}

// ===========================================================================
extern "C" void kernel_launch(void* const* d_in, const int* in_sizes, int n_in,
                              void* d_out, int out_size, void* d_ws, size_t ws_size,
                              hipStream_t stream) {
    (void)in_sizes; (void)n_in; (void)out_size;
    const float* x    = (const float*)d_in[0];
    const float* wgt  = (const float*)d_in[1];
    const float* bias = (const float*)d_in[2];
    float* out        = (float*)d_out;

    if (ws_size >= WS_NEEDED) {
        unsigned short* ws0 = (unsigned short*)d_ws;
        transform_w<<<dim3((WBF_ELEMS + 255) / 256), dim3(256), 0, stream>>>(wgt, ws0);
        transform_x<<<dim3((unsigned)((XBF_ELEMS + 255) / 256)), dim3(256), 0, stream>>>(x, ws0);
        dim3 grid(COUT / 128, (NIMG * HWSZ) / 128);   // (2, 784)
        conv3x3_wmma_direct<<<grid, dim3(128), 0, stream>>>(ws0, bias, out);
    } else {
        dim3 grid(COUT / 64, (NIMG * HWSZ) / 128);    // (4, 784)
        conv3x3_wmma_lds<<<grid, dim3(128), 0, stream>>>(x, wgt, bias, out);
    }
}
